// FlowMambaModel_54820962566288
// MI455X (gfx1250) — matrix operand order repaired and checked by reading, not compiled, not run on hardware
//
#include <hip/hip_runtime.h>
#include <hip/hip_bf16.h>

// ---------------------------------------------------------------------------
// FlowMamba on MI455X (gfx1250, wave32).
// Convs as implicit GEMM via v_wmma_f32_16x16x32_f16 (f32 accumulate).
// State update is the bandwidth-bound part: d-innermost float4 layout,
// coefficients amortized across the 25 velocity copies, double-buffered state.
// ---------------------------------------------------------------------------

typedef __attribute__((ext_vector_type(16))) _Float16 v16h;
typedef __attribute__((ext_vector_type(8)))  float    v8f;

#define HH 32
#define WW 32
#define HW 1024
#define DM 64
#define DS 16
#define NV 25
#define KTOT 576   // 64 in-ch * 9 taps
#define KSL  18    // KTOT / 32

// ---------------------------------------------------------------------------
// Pack OIHW f32 weights into WMMA B-matrix fragments (f16).
// Fragment f = (ntile, kslice). Within a fragment: 32 lanes x 16 f16.
//   lane < 16 : N = ntile*16+lane,  K = ks*32 + {0..7, 16..23}[j]   (j = 0..15)
//   lane >=16 : N = ntile*16+lane-16, K = ks*32 + {8..15, 24..31}[j]
// k enumerates (ci, tap) as k = tap*64 + ci  (tap = ty*3+tx).
// ---------------------------------------------------------------------------
__global__ void k_pack(const float* __restrict__ w, _Float16* __restrict__ wpack,
                       int nOut) {
    int idx = blockIdx.x * blockDim.x + threadIdx.x;
    int total = (nOut / 16) * KSL * 32 * 16;
    if (idx >= total) return;
    int j    = idx & 15;
    int lane = (idx >> 4) & 31;
    int ks   = (idx >> 9) % KSL;
    int nt   = idx / (KSL * 512);
    int k  = ks * 32 + ((lane >> 4) << 3) + (j & 7) + ((j >> 3) << 4);
    int n  = nt * 16 + (lane & 15);
    int ci = k & 63, tap = k >> 6;
    wpack[idx] = (_Float16)w[(n * 64 + ci) * 9 + tap];
}

// ---------------------------------------------------------------------------
// Implicit-GEMM 3x3 circular conv, 64 in-ch, nOut in {16,64}.
//  x16 : activations f16 [HW][64] (pixel-major)
//  out32: f32 at out32[n*sN + p*sP]; out16 (optional): f16 [HW][64]
// One 16-pixel tile per block; one 16-channel N-tile per wave.
// ---------------------------------------------------------------------------
__global__ void k_conv_gemm(const _Float16* __restrict__ x16,
                            const _Float16* __restrict__ wpack,
                            const float* __restrict__ bias,
                            float* __restrict__ out32,
                            _Float16* __restrict__ out16,
                            int sN, int sP, int relu) {
    __shared__ __align__(16) _Float16 tile[16][KTOT];
    const int tid = threadIdx.x;
    const int pt  = blockIdx.x;          // 64 pixel tiles

    // im2col: 16 pixels x 9 taps x 128B (64 f16) = 1152 uint4 chunks
    for (int j = tid; j < 1152; j += blockDim.x) {
        int sub = j & 7;
        int tap = (j >> 3) % 9;
        int pi  = j / 72;
        int p = pt * 16 + pi;
        int y = p >> 5, x = p & 31;
        int ty = tap / 3 - 1, tx = tap % 3 - 1;
        int ps = (((y + ty) & 31) << 5) | ((x + tx) & 31);
        const uint4* src = (const uint4*)(x16 + (size_t)ps * 64) + sub;
        uint4* dst = (uint4*)(&tile[pi][tap * 64]) + sub;
        *dst = *src;
    }
    __syncthreads();

    const int lane  = tid & 31;
    const int wv    = tid >> 5;                 // N-tile owned by this wave
    const int mrow  = lane & 15;                // A-matrix row (pixel in tile)
    const int khalf = (lane >> 4) << 3;         // 0 or 8 (A K-layout)

    v8f acc = {};
    const _Float16* wp = wpack + (size_t)wv * KSL * 32 * 16;

#pragma unroll
    for (int ks = 0; ks < KSL; ++ks) {
        v16h A, Bm;
        // A: lane<16 -> K {0..7, 16..23}; lane>=16 -> K {8..15, 24..31}
        ((uint4*)&A)[0] = *(const uint4*)&tile[mrow][ks * 32 + khalf];
        ((uint4*)&A)[1] = *(const uint4*)&tile[mrow][ks * 32 + 16 + khalf];
        const uint4* bp = (const uint4*)(wp + (size_t)(ks * 32 + lane) * 16);
        ((uint4*)&Bm)[0] = bp[0];
        ((uint4*)&Bm)[1] = bp[1];
        if (ks < KSL - 1)
            __builtin_prefetch(wp + (size_t)((ks + 1) * 32 + lane) * 16, 0, 0);
        acc = __builtin_amdgcn_wmma_f32_16x16x32_f16(
            false, A, false, Bm, (short)0, acc, false, false);
    }

    // D layout: lanes 0-15 -> M=r, N=lane; lanes 16-31 -> M=r+8, N=lane-16
    const int n = wv * 16 + (lane & 15);
    const float b = bias ? bias[n] : 0.0f;
    const int pbase = pt * 16 + ((lane >> 4) << 3);
#pragma unroll
    for (int r = 0; r < 8; ++r) {
        float v = acc[r] + b;
        if (relu) v = fmaxf(v, 0.0f);
        int p = pbase + r;
        out32[(size_t)n * sN + (size_t)p * sP] = v;
        if (out16) out16[(size_t)p * 64 + n] = (_Float16)v;
    }
}

// encoder conv1: 1 -> 64 channels (K=9, no WMMA win), fused ReLU, f16 [HW][64] out
__global__ void k_conv1(const float* __restrict__ x, const float* __restrict__ w1,
                        const float* __restrict__ b1, _Float16* __restrict__ e1) {
    int tid = blockIdx.x * blockDim.x + threadIdx.x;
    if (tid >= DM * HW) return;
    int p = tid & 1023, o = tid >> 10;
    int y = p >> 5, xx = p & 31;
    float acc = b1[o];
#pragma unroll
    for (int tap = 0; tap < 9; ++tap) {
        int ty = tap / 3 - 1, tx = tap % 3 - 1;
        int ps = (((y + ty) & 31) << 5) | ((xx + tx) & 31);
        acc += w1[o * 9 + tap] * x[ps];
    }
    e1[(size_t)p * 64 + o] = (_Float16)fmaxf(acc, 0.0f);
}

// A_bar / B_bar, d-innermost, shared across the 25 velocity copies
__global__ void k_coeff(const float* __restrict__ dconv, const float* __restrict__ logA,
                        const float* __restrict__ Bv, const float* __restrict__ dt_inv,
                        float* __restrict__ abar, float* __restrict__ bbar) {
    int tid = blockIdx.x * blockDim.x + threadIdx.x;
    if (tid >= DM * HW) return;
    int p = tid & 1023, c = tid >> 10;
    float xr = dconv[c * HW + p] + dt_inv[0];
    float delta = (xr > 20.0f) ? xr : log1pf(__expf(xr));
#pragma unroll
    for (int d = 0; d < DS; ++d) {
        float A  = -__expf(logA[c * DS + d]);
        float ab = __expf(delta * A);
        float bb = (ab - 1.0f) / A * Bv[p * DS + d];
        abar[((size_t)c * HW + p) * DS + d] = ab;
        bbar[((size_t)c * HW + p) * DS + d] = bb;
    }
}

// SSM state update + y reduction over d. s layout: [v][c][p][d] (d innermost)
__global__ void k_cell(const float* __restrict__ sPrev, float* __restrict__ sNext,
                       const float* __restrict__ abar, const float* __restrict__ bbar,
                       const float* __restrict__ Cv, const float* __restrict__ u,
                       float* __restrict__ y) {
    int tid = blockIdx.x * blockDim.x + threadIdx.x;
    if (tid >= NV * DM * HW) return;
    int p = tid & 1023;
    int c = (tid >> 10) & 63;
    int v = tid >> 16;
    int vx = v / 5 - 2, vy = v % 5 - 2;           // V_LIST order: x outer, y inner
    int ps = ((((p >> 5) + vy) & 31) << 5) | (((p & 31) + vx) & 31);  // roll(-vy,-vx)
    float uval = u[c * HW + p];
    const float4* sp = (const float4*)(sPrev + ((size_t)(v * 64 + c) * HW + ps) * DS);
    float4*       sn = (float4*)(sNext + ((size_t)(v * 64 + c) * HW + p) * DS);
    const float4* ab = (const float4*)(abar + ((size_t)c * HW + p) * DS);
    const float4* bb = (const float4*)(bbar + ((size_t)c * HW + p) * DS);
    const float4* cv = (const float4*)(Cv + (size_t)p * DS);
    float acc = 0.0f;
#pragma unroll
    for (int q = 0; q < 4; ++q) {
        float4 a = ab[q], b = bb[q], s = sp[q], cc = cv[q];
        float4 ns;
        ns.x = a.x * s.x + b.x * uval;
        ns.y = a.y * s.y + b.y * uval;
        ns.z = a.z * s.z + b.z * uval;
        ns.w = a.w * s.w + b.w * uval;
        acc += ns.x * cc.x + ns.y * cc.y + ns.z * cc.z + ns.w * cc.w;
        sn[q] = ns;
    }
    y[(size_t)(v * 64 + c) * HW + p] = acc;
}

// max over v, add D-skip (constant over v -> commutes with max), emit f16 [HW][64]
__global__ void k_ymax(const float* __restrict__ y, const float* __restrict__ u,
                       const float* __restrict__ Dskip, _Float16* __restrict__ out16) {
    int tid = blockIdx.x * blockDim.x + threadIdx.x;
    if (tid >= DM * HW) return;
    int p = tid & 1023, c = tid >> 10;
    float m = -3.4e38f;
    for (int v = 0; v < NV; ++v)
        m = fmaxf(m, y[(size_t)(v * 64 + c) * HW + p]);
    m += Dskip[c] * u[c * HW + p];
    out16[(size_t)p * 64 + c] = (_Float16)m;
}

// final decoder conv: 64 -> 1 channel; writes prediction frame + feedback buffer
__global__ void k_dec3(const float* __restrict__ d2, const float* __restrict__ w3,
                       const float* __restrict__ b3, float* __restrict__ outp,
                       float* __restrict__ xcur) {
    int p = blockIdx.x * blockDim.x + threadIdx.x;
    if (p >= HW) return;
    int y = p >> 5, x = p & 31;
    float acc = b3[0];
    for (int c = 0; c < DM; ++c) {
#pragma unroll
        for (int tap = 0; tap < 9; ++tap) {
            int ty = tap / 3 - 1, tx = tap % 3 - 1;
            int ps = (((y + ty) & 31) << 5) | ((x + tx) & 31);
            acc += w3[c * 9 + tap] * d2[c * HW + ps];
        }
    }
    outp[p] = acc;
    xcur[p] = acc;
}

// ---------------------------------------------------------------------------
static inline char* carve(char*& p, size_t bytes) {
    char* r = p;
    p += (bytes + 255) & ~(size_t)255;
    return r;
}

extern "C" void kernel_launch(void* const* d_in, const int* in_sizes, int n_in,
                              void* d_out, int out_size, void* d_ws, size_t ws_size,
                              hipStream_t stream) {
    (void)in_sizes; (void)n_in; (void)out_size; (void)ws_size;
    const float* inSeq   = (const float*)d_in[0];   // (1,4,1,32,32)
    const float* enc_w1  = (const float*)d_in[1];
    const float* enc_b1  = (const float*)d_in[2];
    const float* enc_w2  = (const float*)d_in[3];
    const float* enc_b2  = (const float*)d_in[4];
    const float* wd      = (const float*)d_in[5];
    const float* bd      = (const float*)d_in[6];
    const float* wB      = (const float*)d_in[7];
    const float* wC      = (const float*)d_in[8];
    const float* logA    = (const float*)d_in[9];
    const float* Dskip   = (const float*)d_in[10];
    const float* dt_inv  = (const float*)d_in[11];
    const float* dec_w1  = (const float*)d_in[12];
    const float* dec_b1  = (const float*)d_in[13];
    const float* dec_w2  = (const float*)d_in[14];
    const float* dec_b2  = (const float*)d_in[15];
    const float* dec_w3  = (const float*)d_in[16];
    const float* dec_b3  = (const float*)d_in[17];
    float* out = (float*)d_out;                     // (1,4,1,32,32) f32

    // ---- workspace carve ----
    char* wsp = (char*)d_ws;
    const size_t fragB64 = (size_t)4 * KSL * 32 * 16 * sizeof(_Float16);
    const size_t fragB16 = (size_t)1 * KSL * 32 * 16 * sizeof(_Float16);
    _Float16* wpEnc2 = (_Float16*)carve(wsp, fragB64);
    _Float16* wpWd   = (_Float16*)carve(wsp, fragB64);
    _Float16* wpWB   = (_Float16*)carve(wsp, fragB16);
    _Float16* wpWC   = (_Float16*)carve(wsp, fragB16);
    _Float16* wpDec1 = (_Float16*)carve(wsp, fragB64);
    _Float16* wpDec2 = (_Float16*)carve(wsp, fragB64);
    _Float16* e1_16  = (_Float16*)carve(wsp, (size_t)DM * HW * 2);
    float*    u32    = (float*)   carve(wsp, (size_t)DM * HW * 4);
    _Float16* u16    = (_Float16*)carve(wsp, (size_t)DM * HW * 2);
    float*    dconv  = (float*)   carve(wsp, (size_t)DM * HW * 4);
    float*    Bv     = (float*)   carve(wsp, (size_t)DS * HW * 4);
    float*    Cv     = (float*)   carve(wsp, (size_t)DS * HW * 4);
    float*    abar   = (float*)   carve(wsp, (size_t)DM * DS * HW * 4);
    float*    bbar   = (float*)   carve(wsp, (size_t)DM * DS * HW * 4);
    float*    ybuf   = (float*)   carve(wsp, (size_t)NV * DM * HW * 4);
    _Float16* ymax16 = (_Float16*)carve(wsp, (size_t)DM * HW * 2);
    float*    d1_32  = (float*)   carve(wsp, (size_t)DM * HW * 4);
    _Float16* d1_16  = (_Float16*)carve(wsp, (size_t)DM * HW * 2);
    float*    d2_32  = (float*)   carve(wsp, (size_t)DM * HW * 4);
    _Float16* d2_16  = (_Float16*)carve(wsp, (size_t)DM * HW * 2);
    float*    xcur   = (float*)   carve(wsp, (size_t)HW * 4);
    const size_t sBytes = (size_t)NV * DM * HW * DS * 4;  // 104.9 MB
    float*    s0 = (float*)carve(wsp, sBytes);
    float*    s1 = (float*)carve(wsp, sBytes);

    // ---- weight pre-pack (deterministic, re-done every call) ----
    k_pack<<<144, 256, 0, stream>>>(enc_w2, wpEnc2, 64);
    k_pack<<<144, 256, 0, stream>>>(wd,     wpWd,   64);
    k_pack<<<36,  256, 0, stream>>>(wB,     wpWB,   16);
    k_pack<<<36,  256, 0, stream>>>(wC,     wpWC,   16);
    k_pack<<<144, 256, 0, stream>>>(dec_w1, wpDec1, 64);
    k_pack<<<144, 256, 0, stream>>>(dec_w2, wpDec2, 64);

    hipMemsetAsync(s0, 0, sBytes, stream);

    float* sCur = s0;
    float* sNxt = s1;

    // one full "encode + cell" pass given a 1-channel input frame
    auto step_cell = [&](const float* frame) {
        k_conv1<<<256, 256, 0, stream>>>(frame, enc_w1, enc_b1, e1_16);
        // conv2: 64->64, bias + relu, outputs both f32 [C][HW] and f16 [HW][C]
        k_conv_gemm<<<64, 128, 0, stream>>>(e1_16, wpEnc2, enc_b2, u32, u16,
                                            HW, 1, 1);
        // wd: 64->64 + bias (no relu)
        k_conv_gemm<<<64, 128, 0, stream>>>(u16, wpWd, bd, dconv, nullptr,
                                            HW, 1, 0);
        // wB / wC: 64->16, no bias, stored d-innermost [p][d]
        k_conv_gemm<<<64, 32, 0, stream>>>(u16, wpWB, nullptr, Bv, nullptr,
                                           1, DS, 0);
        k_conv_gemm<<<64, 32, 0, stream>>>(u16, wpWC, nullptr, Cv, nullptr,
                                           1, DS, 0);
        k_coeff<<<256, 256, 0, stream>>>(dconv, logA, Bv, dt_inv, abar, bbar);
        k_cell<<<6400, 256, 0, stream>>>(sCur, sNxt, abar, bbar, Cv, u32, ybuf);
        float* t = sCur; sCur = sNxt; sNxt = t;
    };

    // encoding scan over the 4 input frames
    for (int t = 0; t < 4; ++t)
        step_cell(inSeq + (size_t)t * HW);

    // autoregressive decode, pred_len = 4
    const float* frame = inSeq + (size_t)3 * HW;   // last input frame
    for (int stp = 0; stp < 4; ++stp) {
        step_cell(frame);
        k_ymax<<<256, 256, 0, stream>>>(ybuf, u32, Dskip, ymax16);
        k_conv_gemm<<<64, 128, 0, stream>>>(ymax16, wpDec1, dec_b1, d1_32, d1_16,
                                            HW, 1, 1);
        k_conv_gemm<<<64, 128, 0, stream>>>(d1_16, wpDec2, dec_b2, d2_32, d2_16,
                                            HW, 1, 1);
        k_dec3<<<4, 256, 0, stream>>>(d2_32, dec_w3, dec_b3,
                                      out + (size_t)stp * HW, xcur);
        frame = xcur;
    }
}
